// MultiQueryAttention_50130858279278
// MI455X (gfx1250) — compile-verified, matrix operand order checked by reading, not run on hardware
//
#include <hip/hip_runtime.h>

typedef __attribute__((ext_vector_type(16))) __bf16 v16bf;
typedef __attribute__((ext_vector_type(8)))  float  v8f;
typedef __attribute__((ext_vector_type(4))) unsigned int v4u;
typedef __attribute__((ext_vector_type(8))) int v8i_;
typedef __attribute__((ext_vector_type(4))) int v4i_;
typedef unsigned short u16;
typedef unsigned int   u32;

typedef __attribute__((address_space(1))) void as1_void;
typedef __attribute__((address_space(3))) void as3_void;
typedef __attribute__((address_space(1))) v4i_ as1_v4i;   // global int4
typedef __attribute__((address_space(3))) v4i_ as3_v4i;   // LDS int4

#if __has_builtin(__builtin_amdgcn_global_load_async_to_lds_b128)
#define HAVE_ASYNC 1
#else
#define HAVE_ASYNC 0
#endif

#if __has_builtin(__builtin_amdgcn_tensor_load_to_lds)
#define HAVE_TDM 1
#else
#define HAVE_TDM 0
#endif

union AF { v16bf v; uint4 q[2]; };
union CF { v8f   v; float f[8]; };

__device__ __forceinline__ u16 f2bf(float f) {
  u32 u = __builtin_bit_cast(u32, f);
  u += 0x7FFFu + ((u >> 16) & 1u);       // round-to-nearest-even
  return (u16)(u >> 16);
}

__device__ __forceinline__ v8f wmma_bf16(v16bf a, v16bf b, v8f c) {
  // D = A(16x32 bf16) * B(32x16 bf16) + C(16x16 f32)
  return __builtin_amdgcn_wmma_f32_16x16x32_bf16(false, a, false, b,
                                                 (short)0, c, false, false);
}

__device__ __forceinline__ void wait_async0() {
#if HAVE_ASYNC
#if __has_builtin(__builtin_amdgcn_s_wait_asynccnt)
  __builtin_amdgcn_s_wait_asynccnt(0);
#else
  asm volatile("s_wait_asynccnt 0" ::: "memory");
#endif
#endif
}

// Loads a 16x32 bf16 fragment (A layout; also Bt layout when rows are N).
// Per ISA 7.12.2: lane holds row (lane&15); K groups kk+half*8+{0..7} and
// kk+16+half*8+{0..7}  (half = lane>>4). Two contiguous 16-byte reads.
__device__ __forceinline__ AF load_frag(const u16* base, int row, int stride, int kk) {
  const int lane = threadIdx.x & 31;
  const int r    = row + (lane & 15);
  const int half = lane >> 4;
  const u16* p = base + (size_t)r * stride + kk + half * 8;
  AF a;
  a.q[0] = *(const uint4*)(p);
  a.q[1] = *(const uint4*)(p + 16);
  return a;
}

// ---------------------------------------------------------------- convert
__global__ __launch_bounds__(256)
void f32_to_bf16_kernel(const float* __restrict__ in, u16* __restrict__ out, int n) {
  for (int i = blockIdx.x * blockDim.x + threadIdx.x; i < n;
       i += gridDim.x * blockDim.x)
    out[i] = f2bf(in[i]);
}

// ---------------------------------------------------------------- GEMM
// C[M,N] = alpha * A[M,K] @ B[K,N]   (A,B bf16 row-major; C f32 or bf16)
// Block tile 128x64, BK=64, 256 threads = 8 waves in 4x2, each wave 32x32.
template<bool OUT_F32>
__global__ __launch_bounds__(256)
void gemm_ws(const u16* __restrict__ A, const u16* __restrict__ B,
             void* __restrict__ C, int M, int N, int K, float alpha)
{
  constexpr int BM = 128, BN = 64, BK = 64, LDA = BK + 8, LDB = BK + 8;
  __shared__ __align__(16) u16 sA[BM * LDA];   // [m][k]
  __shared__ __align__(16) u16 sBt[BN * LDB];  // [n][k] (transposed)

  const int t = threadIdx.x, lane = t & 31, w = t >> 5;
  const int wm = (w & 3) * 32, wn = (w >> 2) * 32;
  const int m0 = blockIdx.y * BM, n0 = blockIdx.x * BN;

  CF acc[2][2] = {};

  const int ar = t >> 3, acg = (t & 7) * 8;   // A: 32 rows/pass, 8-elt groups
  const int br = t >> 3, bcg = (t & 7) * 8;   // B: 32 k-rows/pass

  for (int k0 = 0; k0 < K; k0 += BK) {
    // ---- A tile: async global->LDS (ASYNCcnt path) when available
#pragma unroll
    for (int p = 0; p < 4; p++) {
      int r = ar + p * 32;
#if HAVE_ASYNC
      __builtin_amdgcn_global_load_async_to_lds_b128(
          (as1_v4i*)(void*)&A[(size_t)(m0 + r) * K + k0 + acg],
          (as3_v4i*)(void*)&sA[r * LDA + acg], 0, 0);
#else
      *(uint4*)&sA[r * LDA + acg] =
          *(const uint4*)&A[(size_t)(m0 + r) * K + k0 + acg];
#endif
    }
    // ---- B tile: transpose through VGPRs (TDM/async cannot transpose)
#pragma unroll
    for (int p = 0; p < 2; p++) {
      int kr = br + p * 32;
      uint4 qv = *(const uint4*)&B[(size_t)(k0 + kr) * N + n0 + bcg];
      const u16* e = (const u16*)&qv;
#pragma unroll
      for (int x = 0; x < 8; x++) sBt[(bcg + x) * LDB + kr] = e[x];
    }
    wait_async0();
    __syncthreads();

    if (k0 + BK < K) {  // gfx1250 global_prefetch_b8 for next tiles
      __builtin_prefetch(&A[(size_t)(m0 + ar) * K + k0 + BK + acg], 0, 1);
      __builtin_prefetch(&B[(size_t)(k0 + BK + br) * N + n0 + bcg], 0, 1);
    }

#pragma unroll
    for (int kk = 0; kk < BK; kk += 32) {
      AF a0 = load_frag(sA,  wm,      LDA, kk);
      AF a1 = load_frag(sA,  wm + 16, LDA, kk);
      AF b0 = load_frag(sBt, wn,      LDB, kk);
      AF b1 = load_frag(sBt, wn + 16, LDB, kk);
      acc[0][0].v = wmma_bf16(a0.v, b0.v, acc[0][0].v);
      acc[0][1].v = wmma_bf16(a0.v, b1.v, acc[0][1].v);
      acc[1][0].v = wmma_bf16(a1.v, b0.v, acc[1][0].v);
      acc[1][1].v = wmma_bf16(a1.v, b1.v, acc[1][1].v);
    }
    __syncthreads();
  }

  const int half = lane >> 4, nl = lane & 15;
#pragma unroll
  for (int i = 0; i < 2; i++)
#pragma unroll
    for (int j = 0; j < 2; j++)
#pragma unroll
      for (int vv = 0; vv < 8; vv++) {
        int m = m0 + wm + i * 16 + vv + half * 8;  // ISA C-layout
        int n = n0 + wn + j * 16 + nl;
        float val = acc[i][j].f[vv] * alpha;
        if (OUT_F32) ((float*)C)[(size_t)m * N + n] = val;
        else         ((u16*)C)[(size_t)m * N + n]   = f2bf(val);
      }
}

// ---------------------------------------------------------------- attention
#define SEQ 2048
#define HD  128
#define NH  16

__global__ __launch_bounds__(256)
void mqa_attn(const u16* __restrict__ Qb,  // [B*S, 2048] bf16 (pre-scaled)
              const u16* __restrict__ Kb,  // [B*S, 128]
              const u16* __restrict__ Vb,  // [B*S, 128]
              u16* __restrict__ Ob)        // [B*S, 2048]
{
  constexpr int LKV = HD + 8;   // 136 u16 = 68 dwords (64 data + 4 pad)
  constexpr int LVT = 64 + 8;   // 72
  __shared__ __align__(16) u16 sK [64 * LKV];       // [key][hd] == Bt for QK^T
  __shared__ __align__(16) u16 sVt[HD * LVT];       // [hd][key] == Bt for PV
  __shared__ __align__(16) u16 sP [8 * 16 * LVT];   // per-wave 16x64 P tile

  const int t = threadIdx.x, lane = t & 31, w = t >> 5;
  const int half = lane >> 4, nl = lane & 15;
  const int qt = blockIdx.x, h = blockIdx.y, b = blockIdx.z;

  const size_t qrow0 = (size_t)b * SEQ + (size_t)qt * 128 + w * 16;

  // Q fragments for this wave's 16 rows: 4 k-steps over HD=128, loaded once
  AF aq[4];
#pragma unroll
  for (int ks = 0; ks < 4; ks++) {
    const u16* p = Qb + (qrow0 + nl) * (NH * HD) + h * HD + ks * 32 + half * 8;
    aq[ks].q[0] = *(const uint4*)p;
    aq[ks].q[1] = *(const uint4*)(p + 16);
  }

  CF o[8] = {};
  float rmax[8], rsum[8];
#pragma unroll
  for (int i = 0; i < 8; i++) { rmax[i] = -3.0e38f; rsum[i] = 0.f; }

  u16* sPw = sP + w * 16 * LVT;

  for (int c0 = 0; c0 < SEQ; c0 += 64) {
    // ---- V chunk: global -> LDS transposed (all threads; needs transpose)
    {
      int kr = t >> 4;            // 0..15
      int cg = (t & 15) * 8;      // hd group
#pragma unroll
      for (int p = 0; p < 4; p++) {
        int key = kr + p * 16;
        uint4 qv = *(const uint4*)&Vb[((size_t)b * SEQ + c0 + key) * HD + cg];
        const u16* e = (const u16*)&qv;
#pragma unroll
        for (int x = 0; x < 8; x++) sVt[(cg + x) * LVT + key] = e[x];
      }
    }
    // ---- K chunk: 64x128 bf16 rectangular tile, natural layout.
#if HAVE_TDM
    // Tensor Data Mover: one wave issues a 2D descriptor; pad_enable
    // reproduces the +4-dword row pad (stride 68 dwords == LKV u16).
    if (w == 0) {
      unsigned lds_base = (unsigned)(size_t)(as3_void*)(void*)&sK[0];
      unsigned long long ga =
          (unsigned long long)(size_t)&Kb[((size_t)b * SEQ + c0) * HD];
      v4u g0 = { 1u,                                   // count=1 (valid D#)
                 lds_base,                             // lds_addr
                 (unsigned)(ga & 0xFFFFFFFFu),         // global_addr lo
                 (unsigned)(((ga >> 32) & 0x01FFFFFFu) | (2u << 30)) }; // hi|type=2
      unsigned d0 = (1u << 16)      // data_size = 2 bytes
                  | (1u << 20)      // pad_enable
                  | (5u << 22)      // pad_interval: 64 dwords (one 128-elt row)
                  | (3u << 25);     // pad_amount: 4 dwords (8 u16)
      v8i_ g1 = { (int)d0,
                  (int)(128u << 16),   // tensor_dim0 = 128 (lo16 @ bits 63:48)
                  (int)(64u  << 16),   // tensor_dim1 = 64
                  (int)(128u << 16),   // tile_dim0 = 128 (bits 127:112)
                  64,                  // tile_dim1 = 64 (bits 143:128)
                  128,                 // tensor_dim0_stride lo32
                  0, 0 };
      v4i_ gz = { 0, 0, 0, 0 };
#if defined(__clang_major__) && __clang_major__ >= 23
      v8i_ gz8 = { 0, 0, 0, 0, 0, 0, 0, 0 };
      __builtin_amdgcn_tensor_load_to_lds(g0, g1, gz, gz, gz8, 0);
#else
      __builtin_amdgcn_tensor_load_to_lds(g0, g1, gz, gz, 0);
#endif
      __builtin_amdgcn_s_wait_tensorcnt(0);
    }
#else
    {
      int kr = t >> 4;
      int cg = (t & 15) * 8;
#pragma unroll
      for (int p = 0; p < 4; p++) {
        int key = kr + p * 16;
        *(uint4*)&sK[key * LKV + cg] =
            *(const uint4*)&Kb[((size_t)b * SEQ + c0 + key) * HD + cg];
      }
    }
#endif
    __syncthreads();

    // scores S = Q(16xHD) @ K^T(HDx64), f32 accum
    CF s[4] = {};
#pragma unroll
    for (int nt = 0; nt < 4; nt++)
#pragma unroll
      for (int ks = 0; ks < 4; ks++) {
        AF bk = load_frag(sK, nt * 16, LKV, ks * 32);
        s[nt].v = wmma_bf16(aq[ks].v, bk.v, s[nt].v);
      }

    // online softmax (rows live in 16-lane halves; butterfly masks 8..1)
#pragma unroll
    for (int i = 0; i < 8; i++) {
      float cm = fmaxf(fmaxf(s[0].f[i], s[1].f[i]), fmaxf(s[2].f[i], s[3].f[i]));
#pragma unroll
      for (int off = 8; off >= 1; off >>= 1) cm = fmaxf(cm, __shfl_xor(cm, off, 32));
      float nm = fmaxf(rmax[i], cm);
      float sf = __expf(rmax[i] - nm);
      rmax[i] = nm;
      float ps = 0.f;
#pragma unroll
      for (int nt = 0; nt < 4; nt++) {
        float pv = __expf(s[nt].f[i] - nm);
        s[nt].f[i] = pv;
        ps += pv;
      }
#pragma unroll
      for (int off = 8; off >= 1; off >>= 1) ps += __shfl_xor(ps, off, 32);
      rsum[i] = rsum[i] * sf + ps;
#pragma unroll
      for (int j = 0; j < 8; j++) o[j].f[i] *= sf;
      // P: C-layout -> per-wave LDS (A-layout staging)
      int row = i + half * 8;
#pragma unroll
      for (int nt = 0; nt < 4; nt++)
        sPw[row * LVT + nt * 16 + nl] = f2bf(s[nt].f[i]);
    }
    asm volatile("s_wait_dscnt 0" ::: "memory");  // wave-local LDS RAW guard

    // O += P(16x64) @ V(64xHD)
    AF ap0 = load_frag(sPw, 0, LVT, 0);
    AF ap1 = load_frag(sPw, 0, LVT, 32);
#pragma unroll
    for (int j = 0; j < 8; j++) {
      AF bv0 = load_frag(sVt, j * 16, LVT, 0);
      AF bv1 = load_frag(sVt, j * 16, LVT, 32);
      o[j].v = wmma_bf16(ap0.v, bv0.v, o[j].v);
      o[j].v = wmma_bf16(ap1.v, bv1.v, o[j].v);
    }
    __syncthreads();
  }

  // normalize and store bf16 [B*S, NH*HD]
#pragma unroll
  for (int i = 0; i < 8; i++) {
    float inv = 1.0f / rsum[i];
    size_t row = qrow0 + i + half * 8;
#pragma unroll
    for (int j = 0; j < 8; j++)
      Ob[row * (NH * HD) + h * HD + j * 16 + nl] = f2bf(o[j].f[i] * inv);
  }
}

// ---------------------------------------------------------------- launch
extern "C" void kernel_launch(void* const* d_in, const int* in_sizes, int n_in,
                              void* d_out, int out_size, void* d_ws, size_t ws_size,
                              hipStream_t stream) {
  const float* x  = (const float*)d_in[0];
  const float* Wq = (const float*)d_in[1];
  const float* Wk = (const float*)d_in[2];
  const float* Wv = (const float*)d_in[3];
  const float* Wo = (const float*)d_in[4];
  float* out = (float*)d_out;

  const size_t NX  = (size_t)2 * 2048 * 2048;  // B*S*D = 8388608
  const size_t NW  = (size_t)2048 * 2048;      // 4194304
  const size_t NKV = (size_t)2048 * 128;       // 262144
  const size_t NKVACT = (size_t)4096 * 128;    // B*S*Hd

  char* p = (char*)d_ws;
  u16* xb  = (u16*)p; p += NX * 2;
  u16* wqb = (u16*)p; p += NW * 2;
  u16* wkb = (u16*)p; p += NKV * 2;
  u16* wvb = (u16*)p; p += NKV * 2;
  u16* wob = (u16*)p; p += NW * 2;
  u16* Qb  = (u16*)p; p += NX * 2;
  u16* Kb  = (u16*)p; p += NKVACT * 2;
  u16* Vb  = (u16*)p; p += NKVACT * 2;
  u16* Ob  = (u16*)p; p += NX * 2;

  auto cvt = [&](const float* in, u16* o, size_t n) {
    int blocks = (int)((n + 255) / 256);
    if (blocks > 2048) blocks = 2048;
    f32_to_bf16_kernel<<<blocks, 256, 0, stream>>>(in, o, (int)n);
  };
  cvt(x,  xb,  NX);
  cvt(Wq, wqb, NW);
  cvt(Wk, wkb, NKV);
  cvt(Wv, wvb, NKV);
  cvt(Wo, wob, NW);

  const float qscale = 0.08838834764831845f;  // 1/sqrt(128), folded into Q
  gemm_ws<false><<<dim3(2048 / 64, 4096 / 128), 256, 0, stream>>>(
      xb, wqb, Qb, 4096, 2048, 2048, qscale);
  gemm_ws<false><<<dim3(128 / 64, 4096 / 128), 256, 0, stream>>>(
      xb, wkb, Kb, 4096, 128, 2048, 1.0f);
  gemm_ws<false><<<dim3(128 / 64, 4096 / 128), 256, 0, stream>>>(
      xb, wvb, Vb, 4096, 128, 2048, 1.0f);

  mqa_attn<<<dim3(16, 16, 2), 256, 0, stream>>>(Qb, Kb, Vb, Ob);

  gemm_ws<true><<<dim3(2048 / 64, 4096 / 128), 256, 0, stream>>>(
      Ob, wob, out, 4096, 2048, 2048, 1.0f);
}
